// MultiHeadSRALayer_46007689675142
// MI455X (gfx1250) — matrix operand order repaired and checked by reading, not compiled, not run on hardware
//
#include <hip/hip_runtime.h>

#define EMBED 4096
#define HEADS 32
#define HDIM  128
#define SEQ   64
#define BATCH 32
#define MROWS (BATCH*SEQ)   // 2048

typedef _Float16 v4h  __attribute__((ext_vector_type(4)));
typedef _Float16 v8h  __attribute__((ext_vector_type(8)));
typedef _Float16 v16h __attribute__((ext_vector_type(16)));
typedef float    v8f  __attribute__((ext_vector_type(8)));

union V16 { v16h v; v8h h[2]; };

// async global -> LDS copy of one 16B chunk (per-lane addresses), ASYNCcnt path
__device__ __forceinline__ void async_cp16(const void* g, unsigned lds) {
  asm volatile("global_load_async_to_lds_b128 %0, %1, off"
               :: "v"(lds), "v"((unsigned long long)(size_t)g) : "memory");
}

// ---------------- f32 -> f16 conversion (4-wide) ----------------
__global__ void sra_cvt_f16(const float* __restrict__ src,
                            _Float16* __restrict__ dst, long n4) {
  long i = (long)blockIdx.x * blockDim.x + threadIdx.x;
  if (i >= n4) return;
  float4 f = ((const float4*)src)[i];
  v4h h = { (_Float16)f.x, (_Float16)f.y, (_Float16)f.z, (_Float16)f.w };
  ((v4h*)dst)[i] = h;
}

// ---------------- WMMA f16 GEMM:  C[m,n] = sum_k A[m,k]*B[n,k] + bias[n] ----
// A: M x K row-major f16 ; B: N x K row-major f16 (computes A @ B^T)
// EPI: 0 = none, 1 = sigmoid
// LAYOUT: 0 = scatter to (b,h,s,d) with m=b*64+s, n=h*128+d ; 1 = row-major MxN
// 128x128 block, BK=32, 3-deep async-LDS pipeline, 4 waves, 4x4 16x16 frags/wave
#define KCH 5   // 4 data chunks (32 halves) + 1 pad chunk per row
template <int EPI, int LAYOUT>
__global__ __launch_bounds__(128) void sra_gemm_f16(
    const _Float16* __restrict__ A, const _Float16* __restrict__ B,
    const float* __restrict__ bias, float* __restrict__ C,
    int M, int N, int K) {
  __shared__ v8h As[3][128 * KCH];   // 30 KB
  __shared__ v8h Bs[3][128 * KCH];   // 30 KB

  const int tid   = threadIdx.x;
  const int lane  = tid & 31;
  const int wid   = tid >> 5;      // 0..3
  const int wm    = wid >> 1;      // 0..1  -> M offset wm*64
  const int wn    = wid & 1;       // 0..1  -> N offset wn*64
  const int l16   = lane & 15;
  const int lhalf = (lane < 16) ? 0 : 1;
  const int m0    = blockIdx.y * 128;
  const int n0    = blockIdx.x * 128;

  v8f acc[4][4];
#pragma unroll
  for (int mi = 0; mi < 4; ++mi)
#pragma unroll
    for (int ni = 0; ni < 4; ++ni)
      acc[mi][ni] = (v8f){0.f,0.f,0.f,0.f,0.f,0.f,0.f,0.f};

  const int T = K >> 5;                  // K-tiles of 32

  // staging: 512 chunks per matrix per tile, 128 threads -> 4 chunks each;
  // 8 async instructions per wave per tile
  auto stage = [&](int t, int buf) {
#pragma unroll
    for (int p = 0; p < 4; ++p) {
      int c   = tid + p * 128;
      int row = c >> 2;
      int kc  = c & 3;
      async_cp16(A + (size_t)(m0 + row) * K + t * 32 + kc * 8,
                 (unsigned)(size_t)&As[buf][row * KCH + kc]);
      async_cp16(B + (size_t)(n0 + row) * K + t * 32 + kc * 8,
                 (unsigned)(size_t)&Bs[buf][row * KCH + kc]);
    }
  };

  auto compute = [&](int buf) {
    V16 af[4], bf[4];
    // A frag (16x32): lane<16 -> M=l16, K {0..7},{16..23}; lane>=16 -> {8..15},{24..31}
#pragma unroll
    for (int mi = 0; mi < 4; ++mi) {
      int r = wm * 64 + mi * 16 + l16;
      af[mi].h[0] = As[buf][r * KCH + lhalf];
      af[mi].h[1] = As[buf][r * KCH + 2 + lhalf];
    }
    // B frag (32x16): lane<16 -> N=l16, K {0..7},{16..23}; lane>=16 -> {8..15},{24..31}
#pragma unroll
    for (int ni = 0; ni < 4; ++ni) {
      int r = wn * 64 + ni * 16 + l16;
      bf[ni].h[0] = Bs[buf][r * KCH + lhalf];
      bf[ni].h[1] = Bs[buf][r * KCH + 2 + lhalf];
    }
#pragma unroll
    for (int mi = 0; mi < 4; ++mi)
#pragma unroll
      for (int ni = 0; ni < 4; ++ni)
        acc[mi][ni] = __builtin_amdgcn_wmma_f32_16x16x32_f16(
            false, af[mi].v, false, bf[ni].v, (short)0, acc[mi][ni],
            false, false);
  };

  // prologue: tiles 0 and 1 in flight
  stage(0, 0);
  stage(1, 1);

  int cur = 0;
  // branch-free steady state: retire tile i (oldest 8 of 16 outstanding),
  // barrier doubles as "tile i visible" + "buf (i+2)%3 free to overwrite"
#pragma unroll 1
  for (int i = 0; i + 2 < T; ++i) {
    asm volatile("s_wait_asynccnt 8" ::: "memory");
    __syncthreads();
    int pre = cur + 2; if (pre >= 3) pre -= 3;
    stage(i + 2, pre);
    compute(cur);
    cur = (cur + 1 == 3) ? 0 : cur + 1;
  }
  // drain: last two tiles
  asm volatile("s_wait_asynccnt 0" ::: "memory");
  __syncthreads();
  compute(cur);
  cur = (cur + 1 == 3) ? 0 : cur + 1;
  compute(cur);

  // epilogue: C/D layout -> VGPR r : M = r + (lane<16 ? 0 : 8), N = lane%16
#pragma unroll
  for (int mi = 0; mi < 4; ++mi) {
    int mbase = m0 + wm * 64 + mi * 16 + ((lane < 16) ? 0 : 8);
#pragma unroll
    for (int ni = 0; ni < 4; ++ni) {
      int n = n0 + wn * 64 + ni * 16 + l16;
      float bval = bias[n];
#pragma unroll
      for (int r = 0; r < 8; ++r) {
        int m = mbase + r;
        float v = acc[mi][ni][r] + bval;
        if (EPI == 1) v = 1.f / (1.f + __expf(-v));
        size_t off;
        if (LAYOUT == 0) {
          int b = m >> 6, s = m & 63;
          int h = n >> 7, d = n & 127;
          off = (((size_t)(b * 32 + h) * 64 + s) << 7) + d;
        } else {
          off = (size_t)m * N + n;
        }
        C[off] = v;
      }
    }
  }
}

// ---------------- RoPE in-place on (b,h,s,d) fp32, paired (d, d+64) ---------
// angle(s,d) = (d & 63) * 10000^(-s/64)   (module's swapped-axis RoPE)
__global__ void sra_rope(float* __restrict__ X, long npairs) {
  long i = (long)blockIdx.x * blockDim.x + threadIdx.x;
  if (i >= npairs) return;
  int dh    = (int)(i & 63);
  long rest = i >> 6;                 // (b*32+h)*64 + s
  int s     = (int)(rest & 63);
  long base = rest << 7;              // *128
  float inv = __expf(-(float)s * 0.14391157f);   // ln(10000)/64
  float ang = (float)dh * inv;
  float sn, c;
  __sincosf(ang, &sn, &c);
  float x0 = X[base + dh];
  float x1 = X[base + dh + 64];
  X[base + dh]      = x0 * c - x1 * sn;
  X[base + dh + 64] = x1 * c + x0 * sn;
}

// ---------------- cumprod over s for each (b,h,d), in place ----------------
__global__ void sra_cumprod(float* __restrict__ G) {
  int i = blockIdx.x * blockDim.x + threadIdx.x;   // (b*32+h)*128 + d
  if (i >= BATCH * HEADS * HDIM) return;
  int d  = i & 127;
  int bh = i >> 7;
  float* p = G + (size_t)bh * SEQ * HDIM + d;
  float prod = 1.f;
  for (int s = 0; s < SEQ; ++s) { prod *= p[s * HDIM]; p[s * HDIM] = prod; }
}

// ---------------- decayed causal scores: S = (Q.K^T)*(gc.(1/gc)^T)/128 * tril
__global__ __launch_bounds__(256) void sra_scores(
    const float* __restrict__ Q, const float* __restrict__ K,
    const float* __restrict__ GC, float* __restrict__ S) {
  __shared__ _Float16 sQ[SEQ * HDIM];   // 16 KB
  __shared__ _Float16 sK[SEQ * HDIM];   // 16 KB
  __shared__ float    sG[SEQ * HDIM];   // 32 KB (fp32: gc spans ~e^-50)
  int bh = blockIdx.x;
  size_t base = (size_t)bh << 13;       // *8192
  int tid = threadIdx.x;
  for (int i = tid; i < SEQ * HDIM; i += 256) {
    sQ[i] = (_Float16)Q[base + i];
    sK[i] = (_Float16)K[base + i];
    sG[i] = GC[base + i];
  }
  __syncthreads();
#pragma unroll 1
  for (int j = 0; j < 16; ++j) {
    int p = j * 256 + tid;
    int q = p >> 6, k = p & 63;
    float r = 0.f;
    if (k <= q) {
      float s1 = 0.f, s2 = 0.f;
      const _Float16* qr = &sQ[q * HDIM];
      const _Float16* kr = &sK[k * HDIM];
      const float* gq = &sG[q * HDIM];
      const float* gk = &sG[k * HDIM];
#pragma unroll 8
      for (int d = 0; d < HDIM; ++d) {
        s1 += (float)qr[d] * (float)kr[d];
        s2 += gq[d] * __builtin_amdgcn_rcpf(gk[d]);
      }
      r = s1 * s2 * 0.0078125f;   // /128
    }
    S[((size_t)bh << 12) + p] = r;
  }
}

// ---------------- out = S @ V  -> f16 (b, s, h, d) row-major 2048x4096 ------
__global__ __launch_bounds__(256) void sra_attn_out(
    const float* __restrict__ S, const float* __restrict__ V,
    _Float16* __restrict__ outh) {
  __shared__ float sS[SEQ * SEQ];       // 16 KB
  int bh  = blockIdx.x;
  int tid = threadIdx.x;
  const float* Sb = S + ((size_t)bh << 12);
  for (int i = tid; i < SEQ * SEQ; i += 256) sS[i] = Sb[i];
  __syncthreads();
  int d  = tid & 127;
  int qh = tid >> 7;                    // 0..1 -> q half
  const float* Vb = V + ((size_t)bh << 13);
  float acc[32];
#pragma unroll
  for (int i = 0; i < 32; ++i) acc[i] = 0.f;
  for (int k = 0; k < SEQ; ++k) {
    float v = Vb[k * HDIM + d];
    const float* sc = &sS[(qh * 32) * SEQ + k];
#pragma unroll
    for (int i = 0; i < 32; ++i) acc[i] += sc[i * SEQ] * v;
  }
  int b = bh >> 5, h = bh & 31;
#pragma unroll
  for (int i = 0; i < 32; ++i) {
    int q = qh * 32 + i;
    outh[(size_t)(b * SEQ + q) * EMBED + h * HDIM + d] = (_Float16)acc[i];
  }
}

extern "C" void kernel_launch(void* const* d_in, const int* in_sizes, int n_in,
                              void* d_out, int out_size, void* d_ws, size_t ws_size,
                              hipStream_t stream) {
  (void)in_sizes; (void)n_in; (void)out_size; (void)ws_size;
  const float* x  = (const float*)d_in[0];
  // d_in[1] is t = arange(SEQ); baked into sra_rope.
  const float* Wq = (const float*)d_in[2];
  const float* bq = (const float*)d_in[3];
  const float* Wk = (const float*)d_in[4];
  const float* bk = (const float*)d_in[5];
  const float* Wv = (const float*)d_in[6];
  const float* bv = (const float*)d_in[7];
  const float* Wd = (const float*)d_in[8];
  const float* bd = (const float*)d_in[9];
  const float* Wo = (const float*)d_in[10];
  const float* bo = (const float*)d_in[11];

  const size_t MK = (size_t)MROWS * EMBED;   // 2048*4096
  const size_t NK = (size_t)EMBED * EMBED;   // 4096*4096

  char* w = (char*)d_ws;
  _Float16* xh  = (_Float16*)w; w += MK * 2;
  _Float16* wqh = (_Float16*)w; w += NK * 2;
  _Float16* wkh = (_Float16*)w; w += NK * 2;
  _Float16* wvh = (_Float16*)w; w += NK * 2;
  _Float16* wdh = (_Float16*)w; w += NK * 2;
  _Float16* woh = (_Float16*)w; w += NK * 2;
  float* Qb = (float*)w; w += MK * 4;
  float* Kb = (float*)w; w += MK * 4;
  float* Vb = (float*)w; w += MK * 4;
  float* Gb = (float*)w; w += MK * 4;
  float* Sb = (float*)w; w += (size_t)BATCH * HEADS * SEQ * SEQ * 4;
  _Float16* ah = (_Float16*)w; w += MK * 2;

  // 1) convert operands to f16
  sra_cvt_f16<<<(int)(MK / 4 / 256), 256, 0, stream>>>(x,  xh,  (long)(MK / 4));
  sra_cvt_f16<<<(int)(NK / 4 / 256), 256, 0, stream>>>(Wq, wqh, (long)(NK / 4));
  sra_cvt_f16<<<(int)(NK / 4 / 256), 256, 0, stream>>>(Wk, wkh, (long)(NK / 4));
  sra_cvt_f16<<<(int)(NK / 4 / 256), 256, 0, stream>>>(Wv, wvh, (long)(NK / 4));
  sra_cvt_f16<<<(int)(NK / 4 / 256), 256, 0, stream>>>(Wd, wdh, (long)(NK / 4));
  sra_cvt_f16<<<(int)(NK / 4 / 256), 256, 0, stream>>>(Wo, woh, (long)(NK / 4));

  // 2) four input projections (WMMA + async-LDS pipeline), scattered to (b,h,s,d)
  dim3 gg(EMBED / 128, MROWS / 128);   // (32,16)
  sra_gemm_f16<0,0><<<gg, 128, 0, stream>>>(xh, wqh, bq, Qb, MROWS, EMBED, EMBED);
  sra_gemm_f16<0,0><<<gg, 128, 0, stream>>>(xh, wkh, bk, Kb, MROWS, EMBED, EMBED);
  sra_gemm_f16<0,0><<<gg, 128, 0, stream>>>(xh, wvh, bv, Vb, MROWS, EMBED, EMBED);
  sra_gemm_f16<1,0><<<gg, 128, 0, stream>>>(xh, wdh, bd, Gb, MROWS, EMBED, EMBED);

  // 3) RoPE on Q,K ; cumprod on gamma
  long npairs = (long)BATCH * HEADS * SEQ * 64;   // 4,194,304
  sra_rope<<<(int)(npairs / 256), 256, 0, stream>>>(Qb, npairs);
  sra_rope<<<(int)(npairs / 256), 256, 0, stream>>>(Kb, npairs);
  sra_cumprod<<<(BATCH * HEADS * HDIM) / 256, 256, 0, stream>>>(Gb);

  // 4) decayed causal attention per (b,h)
  sra_scores<<<BATCH * HEADS, 256, 0, stream>>>(Qb, Kb, Gb, Sb);
  sra_attn_out<<<BATCH * HEADS, 256, 0, stream>>>(Sb, Vb, ah);

  // 5) output projection -> d_out (fp32, row-major 2048x4096)
  sra_gemm_f16<0,1><<<gg, 128, 0, stream>>>(ah, woh, bo, (float*)d_out,
                                            MROWS, EMBED, EMBED);
}
